// AttentionBlock_8693013807796
// MI455X (gfx1250) — matrix operand order repaired and checked by reading, not compiled, not run on hardware
//
#include <hip/hip_runtime.h>
#include <hip/hip_bf16.h>

// ---------------------------------------------------------------------------
// CDNA5 (gfx1250, wave32) attention block:
//   GroupNorm -> q/k/v 1x1 proj -> full spatial softmax attention -> out proj
// All matmuls use v_wmma_f32_16x16x32_bf16 (bf16 operands, f32 accum).
// GEMM A-tiles are staged in LDS (async-to-LDS when available).
// ---------------------------------------------------------------------------

typedef __attribute__((ext_vector_type(16))) __bf16        v16bf;
typedef __attribute__((ext_vector_type(8)))  float         v8f;
typedef __attribute__((ext_vector_type(4)))  unsigned int  u32x4;
typedef __attribute__((ext_vector_type(4)))  int           i32x4;

union V16 {
  v16bf          v;
  u32x4          q[2];
  unsigned short us[16];
};

#if defined(__has_builtin)
#if __has_builtin(__builtin_amdgcn_global_load_async_to_lds_b128)
#define HAS_ASYNC_LDS 1
#endif
#endif
#ifndef HAS_ASYNC_LDS
#define HAS_ASYNC_LDS 0
#endif

#if HAS_ASYNC_LDS
typedef __attribute__((address_space(1))) i32x4 gvec_t;  // global-space v4i
typedef __attribute__((address_space(3))) i32x4 lvec_t;  // LDS-space v4i
#endif

static __device__ __forceinline__ unsigned short f2bf(float f) {
  unsigned int u = __builtin_bit_cast(unsigned int, f);
  u += 0x7fffu + ((u >> 16) & 1u);          // round-to-nearest-even
  return (unsigned short)(u >> 16);
}
static __device__ __forceinline__ float bf2f(unsigned short h) {
  unsigned int u = ((unsigned int)h) << 16;
  return __builtin_bit_cast(float, u);
}
static __device__ __forceinline__ v8f wmma_bf16(v16bf a, v16bf b, v8f c) {
  // (neg_a, A, neg_b, B, c_mod, C, reuse_a, reuse_b)
  return __builtin_amdgcn_wmma_f32_16x16x32_bf16(false, a, false, b, (short)0, c,
                                                 false, false);
}

constexpr int   kB   = 32;
constexpr int   kC   = 512;
constexpr int   kHW  = 1024;   // 32*32 spatial
constexpr int   kG   = 32;     // groups
constexpr int   kCg  = 16;     // channels per group
constexpr float kEps = 1e-5f;
constexpr float kScale = 0.04419417382415922f;  // 1/sqrt(512)

// --------------------------- weight transpose ------------------------------
// WT[o, c] = bf16(W[c, o])   (so GEMM B/A fragments read K=c contiguously)
__global__ __launch_bounds__(256) void wT_bf16_kernel(
    const float* __restrict__ W, unsigned short* __restrict__ WT) {
  int idx = blockIdx.x * 256 + threadIdx.x;
  if (idx >= kC * kC) return;
  int o = idx / kC, c = idx % kC;
  WT[idx] = f2bf(W[c * kC + o]);
}

// ------------------------------ GroupNorm ----------------------------------
// One block per (b, g). Writes hnT[b, s, c] bf16 (spatial-major, K=c contig).
__global__ __launch_bounds__(256) void groupnorm_kernel(
    const float* __restrict__ x, unsigned short* __restrict__ hnT) {
  __shared__ float sm[256], sq[256];
  int b = blockIdx.x / kG, g = blockIdx.x % kG;
  const float* xp = x + ((long)b * kC + (long)g * kCg) * kHW;
  int tid = threadIdx.x;
  float s = 0.f, s2 = 0.f;
  for (int i = tid; i < kCg * kHW; i += 256) {
    float v = xp[i]; s += v; s2 += v * v;
  }
  sm[tid] = s; sq[tid] = s2; __syncthreads();
  for (int ofs = 128; ofs > 0; ofs >>= 1) {
    if (tid < ofs) { sm[tid] += sm[tid + ofs]; sq[tid] += sq[tid + ofs]; }
    __syncthreads();
  }
  const float inv_n = 1.f / (float)(kCg * kHW);
  float mean = sm[0] * inv_n;
  float var  = sq[0] * inv_n - mean * mean;
  float rs   = rsqrtf(var + kEps);
  unsigned short* hp = hnT + (long)b * kHW * kC + g * kCg;
  for (int i = tid; i < kCg * kHW; i += 256) {
    int cl = i >> 10;          // i / kHW
    int sp = i & (kHW - 1);    // i % kHW
    hp[(long)sp * kC + cl] = f2bf((xp[i] - mean) * rs);
  }
}

// ------------------------------- GEMM --------------------------------------
// D[M,N] = A[M,K] * B^T, Bm stored [N,K] row-major (K contiguous), K == 512.
// All 8 waves of a block share one 16-row A tile; it is staged in LDS once
// (async-to-LDS if available), then each wave computes a 16x64 D tile with
// ds_load_b128 A fragments + global_load_b128 B fragments + 4 WMMA per K-step.
// mode 0: bf16 store (row-major [M,N] + batch stride)
// mode 1: f32 store out[z,n,m] = acc + bias[n] + Res[z,n,m]
__global__ __launch_bounds__(256) void gemm_bf16_kernel(
    const unsigned short* __restrict__ A, long strideA,
    const unsigned short* __restrict__ Bm, long strideB,
    const float* __restrict__ bias, int bias_on_m,
    unsigned short* __restrict__ Dbf, long strideD,
    float* __restrict__ Df, const float* __restrict__ Res,
    int M, int N, int K, int mode) {
  __shared__ __align__(16) unsigned short As[16 * 512];  // 16 KB A tile
  int tid = threadIdx.x;
  int wid = tid >> 5, lane = tid & 31;
  int half = lane >> 4, lcol = lane & 15;
  int ntile = N >> 6;         // 16x64 tiles per D row-block
  int tnb = ntile >> 3;       // blocks per tile-row (8 waves per block)
  int tm = blockIdx.x / tnb;
  int tn = (blockIdx.x % tnb) * 8 + wid;
  int z = blockIdx.y;

  // ---- stage the block-shared A tile (16 x K, contiguous 16 KB) ----------
  {
    const u32x4* src = (const u32x4*)(A + (long)z * strideA + (long)(tm * 16) * K);
    u32x4* dst = (u32x4*)As;
    int n128 = (16 * K) >> 3;     // 16-byte chunks
#if HAS_ASYNC_LDS
    for (int i = tid; i < n128; i += 256)
      __builtin_amdgcn_global_load_async_to_lds_b128(
          (gvec_t*)(src + i), (lvec_t*)(dst + i), 0, 0);
#if __has_builtin(__builtin_amdgcn_s_wait_asynccnt)
    __builtin_amdgcn_s_wait_asynccnt(0);
#else
    asm volatile("s_wait_asynccnt 0x0" ::: "memory");
#endif
#else
    for (int i = tid; i < n128; i += 256) dst[i] = src[i];
#endif
  }
  __syncthreads();

  const unsigned short* Als = As + (long)lcol * K;
  const unsigned short* Bb  = Bm + (long)z * strideB;
  const unsigned short* b0p = Bb + (long)(tn * 64 +  0 + lcol) * K;
  const unsigned short* b1p = Bb + (long)(tn * 64 + 16 + lcol) * K;
  const unsigned short* b2p = Bb + (long)(tn * 64 + 32 + lcol) * K;
  const unsigned short* b3p = Bb + (long)(tn * 64 + 48 + lcol) * K;

  v8f acc[4] = {};
  for (int k0 = 0; k0 < K; k0 += 32) {
    V16 a, f0, f1, f2, f3;
    a.q[0] = *(const u32x4*)(Als + k0 + half * 16);
    a.q[1] = *(const u32x4*)(Als + k0 + half * 16 + 8);
    int bo = k0 + half * 16;
    f0.q[0] = *(const u32x4*)(b0p + bo); f0.q[1] = *(const u32x4*)(b0p + bo + 8);
    f1.q[0] = *(const u32x4*)(b1p + bo); f1.q[1] = *(const u32x4*)(b1p + bo + 8);
    f2.q[0] = *(const u32x4*)(b2p + bo); f2.q[1] = *(const u32x4*)(b2p + bo + 8);
    f3.q[0] = *(const u32x4*)(b3p + bo); f3.q[1] = *(const u32x4*)(b3p + bo + 8);
    __builtin_prefetch(b0p + k0 + 128, 0, 1);  // global_prefetch_b8
    acc[0] = wmma_bf16(a.v, f0.v, acc[0]);
    acc[1] = wmma_bf16(a.v, f1.v, acc[1]);
    acc[2] = wmma_bf16(a.v, f2.v, acc[2]);
    acc[3] = wmma_bf16(a.v, f3.v, acc[3]);
  }
#pragma unroll
  for (int j = 0; j < 4; ++j) {
    int ng = tn * 64 + j * 16 + lcol;
#pragma unroll
    for (int r = 0; r < 8; ++r) {
      int mg = tm * 16 + r + half * 8;
      float val = acc[j][r];
      if (bias) val += bias_on_m ? bias[mg] : bias[ng];
      if (mode == 0) {
        Dbf[(long)z * strideD + (long)mg * N + ng] = f2bf(val);
      } else {
        long idx = (long)z * (long)N * M + (long)ng * M + mg;
        Df[idx] = val + Res[idx];
      }
    }
  }
}

// ---------------------------- fused attention ------------------------------
// Per wave: one 16-query tile.  Phase 1 computes S^T (keys x queries) tiles so
// the f32 C-layout (lane = query, rows = keys) matches the bf16 A-operand
// layout of the P*V GEMM exactly -> no cross-lane transpose.  Raw (scaled)
// scores go to LDS as bf16 while max/sum run online; phase 1.5 rewrites LDS as
// normalized exp-probs; phase 2 is a pure WMMA GEMM over 32 channel tiles.
__global__ __launch_bounds__(64) void attention_kernel(
    const unsigned short* __restrict__ kT,    // [B, HW, C]  (A of S^T)
    const unsigned short* __restrict__ qT,    // [B, HW, C]  (B of S^T)
    const unsigned short* __restrict__ v,     // [B, C, HW]  (B of P*V)
    unsigned short* __restrict__ attnT) {     // [B, HW, C]
  __shared__ __align__(16) unsigned short Plds[2 * 16 * 1024];  // 64 KB
  int wid = threadIdx.x >> 5, lane = threadIdx.x & 31;
  int half = lane >> 4, lcol = lane & 15;
  int tile = blockIdx.x * 2 + wid;
  int b  = tile >> 6;
  int s0 = (tile & 63) << 4;
  const unsigned short* kTb = kT + (long)b * kHW * kC;
  const unsigned short* qTb = qT + (long)b * kHW * kC;
  const unsigned short* vb  = v  + (long)b * kC * kHW;
  unsigned short* Pw = Plds + wid * 16 * 1024;
  const unsigned short* qrow = qTb + (long)(s0 + lcol) * kC;  // fixed B column

  float mrun = -3.0e38f, lrun = 0.f;
  for (int chunk = 0; chunk < kHW / 32; ++chunk) {   // 32 keys per chunk
    int t0 = chunk * 32;
    const unsigned short* ka0 = kTb + (long)(t0 + lcol) * kC;
    const unsigned short* ka1 = kTb + (long)(t0 + 16 + lcol) * kC;
    v8f acc0 = {}, acc1 = {};
    for (int k0 = 0; k0 < kC; k0 += 32) {
      V16 a0, a1, bq;
      a0.q[0] = *(const u32x4*)(ka0 + k0 + half * 8);
      a0.q[1] = *(const u32x4*)(ka0 + k0 + 16 + half * 8);
      a1.q[0] = *(const u32x4*)(ka1 + k0 + half * 8);
      a1.q[1] = *(const u32x4*)(ka1 + k0 + 16 + half * 8);
      bq.q[0] = *(const u32x4*)(qrow + k0 + half * 8);
      bq.q[1] = *(const u32x4*)(qrow + k0 + 16 + half * 8);
      acc0 = wmma_bf16(a0.v, bq.v, acc0);
      acc1 = wmma_bf16(a1.v, bq.v, acc1);
    }
    // online softmax stats; lanes L and L^16 hold the same query's key-halves
    float sc[16];
#pragma unroll
    for (int r = 0; r < 8; ++r) {
      sc[r]     = acc0[r] * kScale;
      sc[8 + r] = acc1[r] * kScale;
    }
    float cmax = sc[0];
#pragma unroll
    for (int e = 1; e < 16; ++e) cmax = fmaxf(cmax, sc[e]);
    cmax = fmaxf(cmax, __shfl_xor(cmax, 16, 32));
    float csum = 0.f;
#pragma unroll
    for (int e = 0; e < 16; ++e) csum += __expf(sc[e] - cmax);
    csum += __shfl_xor(csum, 16, 32);
    float nm = fmaxf(mrun, cmax);
    lrun = lrun * __expf(mrun - nm) + csum * __expf(cmax - nm);
    mrun = nm;
    // these 16 values are, in order, the A-operand vector of the P*V chunk
    V16 p;
#pragma unroll
    for (int e = 0; e < 16; ++e) p.us[e] = f2bf(sc[e]);
    u32x4* pp = (u32x4*)Pw + ((long)chunk * 32 + lane) * 2;
    pp[0] = p.q[0]; pp[1] = p.q[1];
  }
  float linv = 1.f / lrun;
  // Phase 1.5: raw scores -> normalized probabilities (in place, per-wave LDS)
  for (int chunk = 0; chunk < kHW / 32; ++chunk) {
    u32x4* pp = (u32x4*)Pw + ((long)chunk * 32 + lane) * 2;
    V16 p; p.q[0] = pp[0]; p.q[1] = pp[1];
#pragma unroll
    for (int e = 0; e < 16; ++e)
      p.us[e] = f2bf(__expf(bf2f(p.us[e]) - mrun) * linv);
    pp[0] = p.q[0]; pp[1] = p.q[1];
  }
  // Phase 2: attnT[s, c] = sum_t P[s,t] * v[c,t]
  for (int ct = 0; ct < kC / 16; ++ct) {
    const unsigned short* vcol = vb + (long)(ct * 16 + lcol) * kHW;
    v8f acc = {};
    for (int chunk = 0; chunk < kHW / 32; ++chunk) {
      u32x4* pp = (u32x4*)Pw + ((long)chunk * 32 + lane) * 2;
      V16 a;  a.q[0] = pp[0]; a.q[1] = pp[1];
      V16 bb;
      int tb = chunk * 32 + half * 8;
      bb.q[0] = *(const u32x4*)(vcol + tb);
      bb.q[1] = *(const u32x4*)(vcol + tb + 16);
      acc = wmma_bf16(a.v, bb.v, acc);
    }
#pragma unroll
    for (int r = 0; r < 8; ++r) {
      attnT[(long)b * kHW * kC + (long)(s0 + r + half * 8) * kC + ct * 16 + lcol] =
          f2bf(acc[r]);
    }
  }
}

// ------------------------------- launcher ----------------------------------
extern "C" void kernel_launch(void* const* d_in, const int* in_sizes, int n_in,
                              void* d_out, int out_size, void* d_ws, size_t ws_size,
                              hipStream_t stream) {
  (void)in_sizes; (void)n_in; (void)out_size; (void)ws_size;
  const float* x  = (const float*)d_in[0];
  const float* Wq = (const float*)d_in[1];
  const float* bq = (const float*)d_in[2];
  const float* Wk = (const float*)d_in[3];
  const float* bk = (const float*)d_in[4];
  const float* Wv = (const float*)d_in[5];
  const float* bv = (const float*)d_in[6];
  const float* Wn = (const float*)d_in[7];
  const float* bn = (const float*)d_in[8];
  float* out = (float*)d_out;

  char* ws = (char*)d_ws;
  const size_t szAct = (size_t)kB * kHW * kC * sizeof(unsigned short); // 32 MiB
  const size_t szW   = (size_t)kC * kC * sizeof(unsigned short);       // 512 KiB
  unsigned short* hnT   = (unsigned short*)(ws);
  unsigned short* qTp   = (unsigned short*)(ws + szAct);
  unsigned short* kTp   = (unsigned short*)(ws + 2 * szAct);
  unsigned short* vP    = (unsigned short*)(ws + 3 * szAct);
  unsigned short* attnT = (unsigned short*)(ws + 4 * szAct);
  unsigned short* WqT   = (unsigned short*)(ws + 5 * szAct);
  unsigned short* WkT   = (unsigned short*)(ws + 5 * szAct + szW);
  unsigned short* WvT   = (unsigned short*)(ws + 5 * szAct + 2 * szW);
  unsigned short* WnT   = (unsigned short*)(ws + 5 * szAct + 3 * szW);

  dim3 b256(256);
  int wtBlocks = (kC * kC) / 256;
  wT_bf16_kernel<<<wtBlocks, b256, 0, stream>>>(Wq, WqT);
  wT_bf16_kernel<<<wtBlocks, b256, 0, stream>>>(Wk, WkT);
  wT_bf16_kernel<<<wtBlocks, b256, 0, stream>>>(Wv, WvT);
  wT_bf16_kernel<<<wtBlocks, b256, 0, stream>>>(Wn, WnT);

  groupnorm_kernel<<<kB * kG, b256, 0, stream>>>(x, hnT);

  dim3 ggrid(64, kB);  // (m_tiles * ntile/8) == 64 for every GEMM below
  // qT[b,s,o] = hnT[b,s,:] . WqT[o,:] + bq[o]
  gemm_bf16_kernel<<<ggrid, b256, 0, stream>>>(
      hnT, (long)kHW * kC, WqT, 0, bq, 0,
      qTp, (long)kHW * kC, nullptr, nullptr, kHW, kC, kC, 0);
  // kT[b,t,o]
  gemm_bf16_kernel<<<ggrid, b256, 0, stream>>>(
      hnT, (long)kHW * kC, WkT, 0, bk, 0,
      kTp, (long)kHW * kC, nullptr, nullptr, kHW, kC, kC, 0);
  // v[b,o,t] = WvT[o,:] . hnT[b,t,:] + bv[o]   (bias over M)
  gemm_bf16_kernel<<<ggrid, b256, 0, stream>>>(
      WvT, 0, hnT, (long)kHW * kC, bv, 1,
      vP, (long)kC * kHW, nullptr, nullptr, kC, kHW, kC, 0);

  attention_kernel<<<(kB * (kHW / 16)) / 2, dim3(64), 0, stream>>>(
      kTp, qTp, vP, attnT);

  // out[b,o,s] = attnT[b,s,:] . WnT[o,:] + bn[o] + x[b,o,s]
  gemm_bf16_kernel<<<ggrid, b256, 0, stream>>>(
      attnT, (long)kHW * kC, WnT, 0, bn, 0,
      nullptr, 0, out, x, kHW, kC, kC, 1);
}
